// GPT_47820165874102
// MI455X (gfx1250) — matrix operand order repaired and checked by reading, not compiled, not run on hardware
//
#include <hip/hip_runtime.h>

#define DEVI __device__ __forceinline__

#ifndef __has_builtin
#define __has_builtin(x) 0
#endif
#if __has_builtin(__builtin_amdgcn_tensor_load_to_lds) && __has_builtin(__builtin_amdgcn_s_wait_tensorcnt)
#define USE_TDM 1
#else
#define USE_TDM 0
#endif

typedef __attribute__((ext_vector_type(16))) __bf16 bf16x16;
typedef __attribute__((ext_vector_type(8)))  float  f32x8;
typedef __attribute__((ext_vector_type(4)))  unsigned int u32x4;
typedef __attribute__((ext_vector_type(8)))  int i32x8;
typedef __attribute__((ext_vector_type(4)))  int i32x4;

union FragBF { uint4 u[2]; bf16x16 v; };

static constexpr int Lnum = 12;
static constexpr int Hn   = 16;
static constexpr int En   = 1024;
static constexpr int Vn   = 8192;
static constexpr int Tn   = 512;
static constexpr int Bn   = 4;
static constexpr int Dn   = 64;          // head dim
static constexpr int BT   = Bn * Tn;     // 2048 rows
static constexpr int BH   = Bn * Hn;     // 64 (b,h) pairs
static constexpr int E4   = 4 * En;      // 4096

// fp32 -> bf16 round-to-nearest-even
DEVI unsigned short f2b(float f) {
  unsigned int u = __float_as_uint(f);
  u += 0x7fffu + ((u >> 16) & 1u);
  return (unsigned short)(u >> 16);
}

DEVI f32x8 zero8() { f32x8 z; for (int i = 0; i < 8; ++i) z[i] = 0.f; return z; }

// ---------------------------------------------------------------- embedding
__global__ __launch_bounds__(256)
void embed_kernel(const int* __restrict__ x, const float* __restrict__ tok,
                  const float* __restrict__ pos, float* __restrict__ h) {
  const int row = blockIdx.x;            // 0..BT-1
  const int t   = row % Tn;
  const int tid = x[row];
  const int e   = threadIdx.x * 4;
  const float4 a = *(const float4*)(tok + (size_t)tid * En + e);
  const float4 p = *(const float4*)(pos + (size_t)t * En + e);
  float4 r; r.x = a.x + p.x; r.y = a.y + p.y; r.z = a.z + p.z; r.w = a.w + p.w;
  *(float4*)(h + (size_t)row * En + e) = r;
}

// ---------------------------------------------------------------- layernorm (fp32 in, bf16 out)
__global__ __launch_bounds__(256)
void ln_kernel(const float* __restrict__ x, const float* __restrict__ w,
               const float* __restrict__ b, unsigned short* __restrict__ y) {
  __shared__ float ss[8], sq[8];
  const int row = blockIdx.x, t = threadIdx.x;
  const int wvi = t >> 5, lane = t & 31;
  const float4 xv = *(const float4*)(x + (size_t)row * En + t * 4);
  float s = xv.x + xv.y + xv.z + xv.w;
  float q = xv.x * xv.x + xv.y * xv.y + xv.z * xv.z + xv.w * xv.w;
#pragma unroll
  for (int m = 16; m; m >>= 1) { s += __shfl_xor(s, m, 32); q += __shfl_xor(q, m, 32); }
  if (lane == 0) { ss[wvi] = s; sq[wvi] = q; }
  __syncthreads();
  if (t < 32) {
    float s2 = (t < 8) ? ss[t] : 0.f;
    float q2 = (t < 8) ? sq[t] : 0.f;
#pragma unroll
    for (int m = 4; m; m >>= 1) { s2 += __shfl_xor(s2, m, 32); q2 += __shfl_xor(q2, m, 32); }
    if (t == 0) { ss[0] = s2; sq[0] = q2; }
  }
  __syncthreads();
  const float mean = ss[0] * (1.f / En);
  const float var  = sq[0] * (1.f / En) - mean * mean;
  const float rstd = rsqrtf(var + 1e-5f);
  const float4 wv4 = *(const float4*)(w + t * 4);
  const float4 bv4 = *(const float4*)(b + t * 4);
  ushort4 o;
  o.x = f2b((xv.x - mean) * rstd * wv4.x + bv4.x);
  o.y = f2b((xv.y - mean) * rstd * wv4.y + bv4.y);
  o.z = f2b((xv.z - mean) * rstd * wv4.z + bv4.z);
  o.w = f2b((xv.w - mean) * rstd * wv4.w + bv4.w);
  *(ushort4*)(y + (size_t)row * En + t * 4) = o;
}

// ---------------------------------------------------------------- generic WMMA GEMM
// C[M,N] = A_bf16[M,K] @ W_fp32[K,N] (+bias). Epilogues:
//   0: fp32 store   1: fp32 residual add   2: GELU -> bf16   3: bf16 store
// A tile (128x32 bf16, row stride K) is staged via the Tensor Data Mover when
// available (wave 0 issues the 2-D D#, waits on TENSORcnt, block barrier
// publishes); W tile needs fp32->bf16 conversion so it stays on the VALU path.
template <int EPI>
__global__ __launch_bounds__(256)
void gemm_kernel(const unsigned short* __restrict__ A, const float* __restrict__ W,
                 const float* __restrict__ bias, const float* res,
                 float* outF, unsigned short* outH, int M, int N, int K) {
  __shared__ unsigned short As[128 * 32];   // [m][k] bf16, 8 KB
  __shared__ unsigned short Wt[128 * 32];   // [n][k] bf16 (transposed), 8 KB
  const int t  = threadIdx.x;
  const int n0 = blockIdx.x * 128, m0 = blockIdx.y * 128;
  const int wv = t >> 5, lane = t & 31, hi = lane >> 4, ln = lane & 15;
  const int wm = (wv & 1) * 64, wn = (wv >> 1) * 32;   // 2x4 wave grid -> 64x32 per wave

  f32x8 acc[4][2];
#pragma unroll
  for (int mi = 0; mi < 4; ++mi)
#pragma unroll
    for (int ni = 0; ni < 2; ++ni) acc[mi][ni] = zero8();

  for (int k0 = 0; k0 < K; k0 += 32) {
    __syncthreads();
#if USE_TDM
    if (wv == 0) {
      // Tensor DMA descriptor: 2-D tile, 128 rows x 32 cols of 2-byte elems.
      const unsigned long long ga =
          (unsigned long long)(const void*)(A + (size_t)m0 * K + k0);
      const unsigned int lds = (unsigned int)(size_t)(void*)As;
      u32x4 g0 = {
        1u,                                               // count=1 (valid user D#)
        lds,                                              // lds_addr (bytes)
        (unsigned int)ga,                                 // global_addr[31:0]
        ((unsigned int)(ga >> 32) & 0x1FFFFFFu) | (2u << 30) // addr[56:32] | type=2
      };
      i32x8 g1 = {
        1 << 16,                                          // data_size=2B, wg_mask=0
        (K & 0xFFFF) << 16,                               // tensor_dim0[15:0]
        (K >> 16) | ((M & 0xFFFF) << 16),                 // dim0[31:16] | tensor_dim1[15:0]
        (M >> 16) | (32 << 16),                           // dim1[31:16] | tile_dim0=32
        128,                                              // tile_dim1=128, tile_dim2=0
        K,                                                // tensor_dim0_stride[31:0]
        0, 0                                              // stride hi / dim1 stride unused
      };
      i32x4 gz = {0, 0, 0, 0};
      i32x8 gz8 = {0, 0, 0, 0, 0, 0, 0, 0};
      __builtin_amdgcn_tensor_load_to_lds(g0, g1, gz, gz, gz8, 0);
      __builtin_amdgcn_s_wait_tensorcnt((short)0);
    }
#else
    // manual A tile staging (bf16, 16B vector loads)
#pragma unroll
    for (int it = 0; it < 2; ++it) {
      const int c = t + it * 256;
      const int r = c >> 2, co = (c & 3) * 8;
      const uint4 u = *(const uint4*)(A + (size_t)(m0 + r) * K + k0 + co);
      *(uint4*)(&As[r * 32 + co]) = u;
    }
#endif
    // stage W tile: fp32 -> bf16, transposed to [n][k]
#pragma unroll
    for (int it = 0; it < 4; ++it) {
      const int c  = t + it * 256;
      const int kk = c >> 5, ng = (c & 31) * 4;
      const float4 w4 = *(const float4*)(W + (size_t)(k0 + kk) * N + n0 + ng);
      Wt[(ng + 0) * 32 + kk] = f2b(w4.x);
      Wt[(ng + 1) * 32 + kk] = f2b(w4.y);
      Wt[(ng + 2) * 32 + kk] = f2b(w4.z);
      Wt[(ng + 3) * 32 + kk] = f2b(w4.w);
    }
    if (k0 + 32 < K)   // warm L2 for next W tile (global_prefetch_b8)
      __builtin_prefetch(W + (size_t)(k0 + 32 + (t >> 3)) * N + n0 + (t & 7) * 16, 0, 0);
    __syncthreads();

    FragBF bfr[2];
#pragma unroll
    for (int ni = 0; ni < 2; ++ni) {
      const unsigned short* bp = &Wt[(wn + 16 * ni + ln) * 32 + 16 * hi];
      bfr[ni].u[0] = *(const uint4*)bp;
      bfr[ni].u[1] = *(const uint4*)(bp + 8);
    }
#pragma unroll
    for (int mi = 0; mi < 4; ++mi) {
      const unsigned short* ap = &As[(wm + 16 * mi + ln) * 32];
      FragBF a;
      a.u[0] = *(const uint4*)(ap + 16 * hi);
      a.u[1] = *(const uint4*)(ap + 16 * hi + 8);
#pragma unroll
      for (int ni = 0; ni < 2; ++ni)
        acc[mi][ni] = __builtin_amdgcn_wmma_f32_16x16x32_bf16(
            false, a.v, false, bfr[ni].v, (short)0, acc[mi][ni], false, false);
    }
  }

#pragma unroll
  for (int mi = 0; mi < 4; ++mi)
#pragma unroll
    for (int ni = 0; ni < 2; ++ni) {
      const int n = n0 + wn + 16 * ni + ln;
      const float bvv = bias ? bias[n] : 0.f;
#pragma unroll
      for (int r = 0; r < 8; ++r) {
        const int m = m0 + wm + 16 * mi + 8 * hi + r;
        const size_t idx = (size_t)m * N + n;
        const float vl = acc[mi][ni][r] + bvv;
        if (EPI == 0)      outF[idx] = vl;
        else if (EPI == 1) outF[idx] = res[idx] + vl;
        else if (EPI == 2) outH[idx] = f2b(0.5f * vl * (1.f + erff(vl * 0.70710678118f)));
        else               outH[idx] = f2b(vl);
      }
    }
}

// ---------------------------------------------------------------- attention scores: S = scale * K Q^T, causal mask
__global__ __launch_bounds__(256)
void attn_scores_kernel(const unsigned short* __restrict__ Kb,
                        const unsigned short* __restrict__ Qb,
                        float* __restrict__ S) {
  const int t  = threadIdx.x;
  const int j0 = blockIdx.x * 128, i0 = blockIdx.y * 128;
  const int bh = blockIdx.z, b = bh >> 4, h = bh & 15;
  const int wv = t >> 5, lane = t & 31, hi = lane >> 4, ln = lane & 15;
  const int wm = (wv & 1) * 64, wn = (wv >> 1) * 32;

  f32x8 acc[4][2];
#pragma unroll
  for (int mi = 0; mi < 4; ++mi)
#pragma unroll
    for (int ni = 0; ni < 2; ++ni) acc[mi][ni] = zero8();

  if (j0 <= i0 + 127) {                       // uniform per-block: skip fully-masked tiles
    const unsigned short* kp = Kb + ((size_t)b * Tn) * En + h * Dn;
    const unsigned short* qp = Qb + ((size_t)b * Tn) * En + h * Dn;
#pragma unroll
    for (int ks = 0; ks < 2; ++ks) {          // D = 64 -> two k-steps of 32
      FragBF bfr[2];
#pragma unroll
      for (int ni = 0; ni < 2; ++ni) {
        const unsigned short* bp = qp + (size_t)(j0 + wn + 16 * ni + ln) * En + ks * 32 + 16 * hi;
        bfr[ni].u[0] = *(const uint4*)bp;
        bfr[ni].u[1] = *(const uint4*)(bp + 8);
      }
#pragma unroll
      for (int mi = 0; mi < 4; ++mi) {
        const unsigned short* ap = kp + (size_t)(i0 + wm + 16 * mi + ln) * En + ks * 32;
        FragBF a;
        a.u[0] = *(const uint4*)(ap + 16 * hi);
        a.u[1] = *(const uint4*)(ap + 16 * hi + 8);
#pragma unroll
        for (int ni = 0; ni < 2; ++ni)
          acc[mi][ni] = __builtin_amdgcn_wmma_f32_16x16x32_bf16(
              false, a.v, false, bfr[ni].v, (short)0, acc[mi][ni], false, false);
      }
    }
  }

  float* sp = S + (size_t)bh * Tn * Tn;
  const float NEG = -__builtin_inff();
#pragma unroll
  for (int mi = 0; mi < 4; ++mi)
#pragma unroll
    for (int ni = 0; ni < 2; ++ni)
#pragma unroll
      for (int r = 0; r < 8; ++r) {
        const int ig = i0 + wm + 16 * mi + 8 * hi + r;
        const int jg = j0 + wn + 16 * ni + ln;
        sp[(size_t)ig * Tn + jg] = (jg <= ig) ? acc[mi][ni][r] * 0.03125f : NEG;  // scale = E^-0.5
      }
}

// ---------------------------------------------------------------- softmax rows of S -> bf16 P
__global__ __launch_bounds__(256)
void softmax_kernel(const float* __restrict__ S, unsigned short* __restrict__ P) {
  const int wv = threadIdx.x >> 5, lane = threadIdx.x & 31;
  const size_t ri = (size_t)blockIdx.x * 8 + wv;   // row over [BH*T)
  const float* sp = S + ri * Tn;
  float xs[16];
  float mx = -__builtin_inff();
#pragma unroll
  for (int c = 0; c < 16; ++c) { xs[c] = sp[c * 32 + lane]; mx = fmaxf(mx, xs[c]); }
#pragma unroll
  for (int m = 16; m; m >>= 1) mx = fmaxf(mx, __shfl_xor(mx, m, 32));
  float sum = 0.f;
#pragma unroll
  for (int c = 0; c < 16; ++c) { xs[c] = expf(xs[c] - mx); sum += xs[c]; }
#pragma unroll
  for (int m = 16; m; m >>= 1) sum += __shfl_xor(sum, m, 32);
  const float inv = 1.f / sum;
  unsigned short* pp = P + ri * Tn;
#pragma unroll
  for (int c = 0; c < 16; ++c) pp[c * 32 + lane] = f2b(xs[c] * inv);
}

// ---------------------------------------------------------------- O = P @ V, write [B,T,E]-layout bf16
__global__ __launch_bounds__(256)
void attn_pv_kernel(const unsigned short* __restrict__ P,
                    const unsigned short* __restrict__ Vb,
                    unsigned short* __restrict__ O) {
  __shared__ unsigned short Vt[64 * 32];      // V^T tile: [d][k], 4 KB
  const int t  = threadIdx.x;
  const int i0 = blockIdx.x * 128;
  const int bh = blockIdx.y, b = bh >> 4, h = bh & 15;
  const int wv = t >> 5, lane = t & 31, hi = lane >> 4, ln = lane & 15;

  f32x8 acc[4];
#pragma unroll
  for (int ni = 0; ni < 4; ++ni) acc[ni] = zero8();

  const unsigned short* pp = P + (size_t)bh * Tn * Tn;
  for (int k0 = 0; k0 < Tn; k0 += 32) {
    __syncthreads();
#pragma unroll
    for (int it = 0; it < 8; ++it) {
      const int c = t + it * 256;
      const int d = c & 63, kk = c >> 6;
      Vt[d * 32 + kk] = Vb[((size_t)b * Tn + k0 + kk) * En + h * Dn + d];
    }
    __syncthreads();
    FragBF a;
    const unsigned short* ap = pp + (size_t)(i0 + 16 * wv + ln) * Tn + k0;
    a.u[0] = *(const uint4*)(ap + 16 * hi);
    a.u[1] = *(const uint4*)(ap + 16 * hi + 8);
#pragma unroll
    for (int ni = 0; ni < 4; ++ni) {
      const unsigned short* bp = &Vt[(16 * ni + ln) * 32 + 16 * hi];
      FragBF bf;
      bf.u[0] = *(const uint4*)bp;
      bf.u[1] = *(const uint4*)(bp + 8);
      acc[ni] = __builtin_amdgcn_wmma_f32_16x16x32_bf16(
          false, a.v, false, bf.v, (short)0, acc[ni], false, false);
    }
  }
#pragma unroll
  for (int ni = 0; ni < 4; ++ni)
#pragma unroll
    for (int r = 0; r < 8; ++r) {
      const int i = i0 + 16 * wv + 8 * hi + r;
      const int d = 16 * ni + ln;
      O[((size_t)b * Tn + i) * En + h * Dn + d] = f2b(acc[ni][r]);
    }
}

// ---------------------------------------------------------------- host-side orchestration
extern "C" void kernel_launch(void* const* d_in, const int* in_sizes, int n_in,
                              void* d_out, int out_size, void* d_ws, size_t ws_size,
                              hipStream_t stream) {
  (void)in_sizes; (void)n_in; (void)out_size; (void)ws_size;
  const int*   x    = (const int*)  d_in[0];
  const float* tok  = (const float*)d_in[1];
  const float* pos  = (const float*)d_in[2];
  const float* Wq   = (const float*)d_in[3];
  const float* bq   = (const float*)d_in[4];
  const float* Wk   = (const float*)d_in[5];
  const float* bk   = (const float*)d_in[6];
  const float* Wv   = (const float*)d_in[7];
  const float* bvp  = (const float*)d_in[8];
  const float* Wo   = (const float*)d_in[9];
  const float* bo   = (const float*)d_in[10];
  const float* ln1w = (const float*)d_in[11];
  const float* ln1b = (const float*)d_in[12];
  const float* ln2w = (const float*)d_in[13];
  const float* ln2b = (const float*)d_in[14];
  const float* W1   = (const float*)d_in[15];
  const float* b1   = (const float*)d_in[16];
  const float* W2   = (const float*)d_in[17];
  const float* b2   = (const float*)d_in[18];
  const float* lnfw = (const float*)d_in[19];
  const float* lnfb = (const float*)d_in[20];
  const float* headW = (const float*)d_in[21];

  char* wp = (char*)d_ws;
  auto alloc = [&](size_t bytes) { char* p = wp; wp += (bytes + 255) & ~(size_t)255; return p; };
  float*          h  = (float*)         alloc((size_t)BT * En * 4);      //  8 MB residual stream
  unsigned short* y  = (unsigned short*)alloc((size_t)BT * En * 2);      //  4 MB LN output (bf16)
  unsigned short* qb = (unsigned short*)alloc((size_t)BT * En * 2);
  unsigned short* kb = (unsigned short*)alloc((size_t)BT * En * 2);
  unsigned short* vb = (unsigned short*)alloc((size_t)BT * En * 2);
  float*          S  = (float*)         alloc((size_t)BH * Tn * Tn * 4); // 64 MB scores
  unsigned short* P  = (unsigned short*)alloc((size_t)BH * Tn * Tn * 2); // 32 MB probs (bf16)
  unsigned short* ao = (unsigned short*)alloc((size_t)BT * En * 2);      //  4 MB attn out (bf16)
  unsigned short* mh = (unsigned short*)alloc((size_t)BT * E4 * 2);      // 16 MB MLP hidden (bf16)

  const dim3 blk(256);
  embed_kernel<<<BT, blk, 0, stream>>>(x, tok, pos, h);
  for (int l = 0; l < Lnum; ++l) {
    ln_kernel<<<BT, blk, 0, stream>>>(h, ln1w + (size_t)l * En, ln1b + (size_t)l * En, y);
    gemm_kernel<3><<<dim3(En / 128, BT / 128), blk, 0, stream>>>(
        y, Wq + (size_t)l * En * En, bq + (size_t)l * En, nullptr, nullptr, qb, BT, En, En);
    gemm_kernel<3><<<dim3(En / 128, BT / 128), blk, 0, stream>>>(
        y, Wk + (size_t)l * En * En, bk + (size_t)l * En, nullptr, nullptr, kb, BT, En, En);
    gemm_kernel<3><<<dim3(En / 128, BT / 128), blk, 0, stream>>>(
        y, Wv + (size_t)l * En * En, bvp + (size_t)l * En, nullptr, nullptr, vb, BT, En, En);
    attn_scores_kernel<<<dim3(Tn / 128, Tn / 128, BH), blk, 0, stream>>>(kb, qb, S);
    softmax_kernel<<<dim3(BH * Tn / 8), blk, 0, stream>>>(S, P);
    attn_pv_kernel<<<dim3(Tn / 128, BH), blk, 0, stream>>>(P, vb, ao);
    gemm_kernel<1><<<dim3(En / 128, BT / 128), blk, 0, stream>>>(
        ao, Wo + (size_t)l * En * En, bo + (size_t)l * En, h, h, nullptr, BT, En, En);
    ln_kernel<<<BT, blk, 0, stream>>>(h, ln2w + (size_t)l * En, ln2b + (size_t)l * En, y);
    gemm_kernel<2><<<dim3(E4 / 128, BT / 128), blk, 0, stream>>>(
        y, W1 + (size_t)l * En * E4, b1 + (size_t)l * E4, nullptr, nullptr, mh, BT, E4, En);
    gemm_kernel<1><<<dim3(En / 128, BT / 128), blk, 0, stream>>>(
        mh, W2 + (size_t)l * E4 * En, b2 + (size_t)l * En, h, h, nullptr, BT, En, E4);
  }
  ln_kernel<<<BT, blk, 0, stream>>>(h, lnfw, lnfb, y);
  gemm_kernel<0><<<dim3(Vn / 128, BT / 128), blk, 0, stream>>>(
      y, headW, nullptr, nullptr, (float*)d_out, nullptr, BT, Vn, En);
}
